// DGM_d_1657857376407
// MI455X (gfx1250) — matrix operand order repaired and checked by reading, not compiled, not run on hardware
//
#include <hip/hip_runtime.h>
#include <hip/hip_bf16.h>
#include <math.h>

typedef __attribute__((ext_vector_type(16))) _Float16 v16h;
typedef __attribute__((ext_vector_type(8)))  float    v8f;
typedef __attribute__((ext_vector_type(4)))  float    v4f;

constexpr int  B_ = 4, N_ = 4096, D_ = 128, K_ = 16;
constexpr int  TILE = 16;
constexpr int  WAVES = 8;                         // 256 threads / wave32
constexpr int  JT_PER_WAVE = (N_ / TILE) / WAVES; // 32 column tiles per wave
constexpr float LN2 = 0.69314718055994531f;

// ---------------- mean over N (per batch, per feature) ----------------
__global__ void k_mean(const float* __restrict__ x, float* __restrict__ mean) {
  const int b = blockIdx.x;
  const int d = threadIdx.x;                      // 128 threads = D_
  const float* xb = x + (size_t)b * N_ * D_ + d;
  float s = 0.f;
  for (int n = 0; n < N_; ++n) s += xb[(size_t)n * D_];
  mean[b * D_ + d] = s * (1.0f / N_);
}

// ------------- center, convert to f16, row squared-norms --------------
__global__ void k_center(const float* __restrict__ x, const float* __restrict__ mean,
                         _Float16* __restrict__ xch, float* __restrict__ sq) {
  const int row = blockIdx.x;                     // b*N + n
  const int b = row / N_;
  const int d = threadIdx.x;                      // 128 threads
  float xc = x[(size_t)row * D_ + d] - mean[b * D_ + d];
  xch[(size_t)row * D_ + d] = (_Float16)xc;
  float v = xc * xc;
  #pragma unroll
  for (int off = 16; off > 0; off >>= 1) v += __shfl_xor(v, off, 32);
  __shared__ float part[4];
  const int w = threadIdx.x >> 5;
  if ((threadIdx.x & 31) == 0) part[w] = v;
  __syncthreads();
  if (threadIdx.x == 0) sq[row] = part[0] + part[1] + part[2] + part[3];
}

// ---------------- fused Gram(WMMA) + gumbel + top-k -------------------
__global__ void __launch_bounds__(256) k_main(
    const _Float16* __restrict__ xch, const float* __restrict__ sq,
    const float* __restrict__ q, const float* __restrict__ temp,
    float* __restrict__ out) {
  __shared__ float ts[WAVES][TILE][TILE + 4];   // stride 20: 16B-aligned f4 rows
  __shared__ float mv[TILE][256];               // merge candidates (values)
  __shared__ int   mi[TILE][256];               // merge candidates (indices)

  const int b    = blockIdx.x / (N_ / TILE);
  const int i0   = (blockIdx.x % (N_ / TILE)) * TILE;
  const int tid  = threadIdx.x;
  const int wave = tid >> 5;
  const int lane = tid & 31;
  const int half = lane >> 4;                   // 0: lanes 0-15, 1: lanes 16-31
  const int lm   = lane & 15;

  float t = temp[0];
  t = fminf(fmaxf(t, -4.0f), 4.0f);
  const float expT = __expf(t);

  const _Float16* xb  = xch + (size_t)b * N_ * D_;
  const float*    sqb = sq  + (size_t)b * N_;

  // Preload A fragments for the 16-row i-tile (16-bit A 16x32 layout, 4 K-chunks)
  v16h afrag[4];
  #pragma unroll
  for (int c = 0; c < 4; ++c) {
    #pragma unroll
    for (int v = 0; v < 8; ++v) {
      const int k = 32 * c + 16 * half + 2 * v;
      const _Float16* p = xb + (size_t)(i0 + lm) * D_ + k;
      afrag[c][2 * v]     = p[0];
      afrag[c][2 * v + 1] = p[1];
    }
  }
  const float sqi = sqb[i0 + lm];

  // Per-lane sorted (descending) top-16 of the columns this lane observes
  float tv[16]; int ti[16];
  #pragma unroll
  for (int k = 0; k < 16; ++k) { tv[k] = -3.3e38f; ti[k] = 0; }

  const int r  = lm;                            // row this lane post-processes
  const int c8 = 8 * half;                      // its 8-column window

  for (int jt = 0; jt < JT_PER_WAVE; ++jt) {
    const int j0 = (jt * WAVES + wave) * TILE;

    // --- issue streaming loads early: q (nontemporal, read-once) and sq ---
    const float* qrow = q + ((size_t)(b * N_ + i0 + r)) * (size_t)N_ + j0 + c8;
    const v4f qv0 = __builtin_nontemporal_load((const v4f*)qrow);
    const v4f qv1 = __builtin_nontemporal_load((const v4f*)(qrow + 4));
    const v4f sj0 = *(const v4f*)(sqb + j0 + c8);
    const v4f sj1 = *(const v4f*)(sqb + j0 + c8 + 4);

    // --- load ALL B fragments (4 K-chunks) so the 8 b128 loads overlap ---
    v16h bf[4];
    #pragma unroll
    for (int c = 0; c < 4; ++c) {
      #pragma unroll
      for (int v = 0; v < 8; ++v) {
        const int k = 32 * c + 16 * half + 2 * v;   // B = xc_j^T, contiguous pairs
        const _Float16* p = xb + (size_t)(j0 + lm) * D_ + k;
        bf[c][2 * v]     = p[0];
        bf[c][2 * v + 1] = p[1];
      }
    }

    // --- prefetch next tile's B rows toward the WGP (L2-resident source) ---
    if (jt + 1 < JT_PER_WAVE) {
      const int j0n = ((jt + 1) * WAVES + wave) * TILE;
      const _Float16* pn = xb + (size_t)(j0n + lm) * D_;
      __builtin_prefetch(pn, 0, 1);
      __builtin_prefetch(pn + 64, 0, 1);
    }

    // --- 4 back-to-back WMMAs, K = 128 ---
    v8f acc = {};
    #pragma unroll
    for (int c = 0; c < 4; ++c)
      acc = __builtin_amdgcn_wmma_f32_16x16x32_f16(
          false, afrag[c], false, bf[c], (short)0, acc, false, false);

    // Transpose 16x16 tile through per-wave LDS (same-wave DS ops are in-order)
    #pragma unroll
    for (int v = 0; v < 8; ++v) ts[wave][v + 8 * half][lm] = acc[v];
    asm volatile("s_wait_dscnt 0x0" ::: "memory");

    const v4f d0 = *(const v4f*)&ts[wave][r][c8];
    const v4f d1 = *(const v4f*)&ts[wave][r][c8 + 4];

    #pragma unroll
    for (int cc = 0; cc < 8; ++cc) {
      const float dot = (cc < 4) ? d0[cc & 3] : d1[cc & 3];
      const float sjj = (cc < 4) ? sj0[cc & 3] : sj1[cc & 3];
      const float qq  = (cc < 4) ? qv0[cc & 3] : qv1[cc & 3];
      const int   j   = j0 + c8 + cc;
      float dd = sqi + sjj - 2.0f * dot;
      dd = fmaxf(dd, 0.0f);
      if (j == i0 + r) dd += 1e20f;             // mask self-edge
      // gumbel: -ln(-ln(q+eps)) via bare v_log_f32 (log2), inputs in safe range
      const float u  = __builtin_amdgcn_logf(qq + 1e-8f) * (-LN2); // -ln(q+eps) > 0
      const float g  = __builtin_amdgcn_logf(u) * LN2;             //  ln(-ln(q+eps))
      const float lq = -dd * expT - g;
      if (lq > tv[15]) {                        // guarded register insertion sort
        float cv2 = lq; int cj = j;
        #pragma unroll
        for (int k = 0; k < 16; ++k) {
          const bool  sw   = cv2 > tv[k];
          const float ftmp = tv[k];
          const int   itmp = ti[k];
          tv[k] = sw ? cv2 : ftmp;  ti[k] = sw ? cj : itmp;
          cv2   = sw ? ftmp : cv2;  cj    = sw ? itmp : cj;
        }
      }
    }
  }

  // Dump per-lane candidates: row lm gets 256 candidates (8 waves x 2 halves x 16)
  const int base = wave * 32 + half * 16;
  #pragma unroll
  for (int k = 0; k < 16; ++k) { mv[lm][base + k] = tv[k]; mi[lm][base + k] = ti[k]; }
  __syncthreads();

  // Final extraction: wave w handles rows 2w, 2w+1; emits sorted-descending top-16
  const size_t E0  = (size_t)B_ * N_ * D_;        // edges offset in d_out
  const size_t BNK = (size_t)B_ * N_ * K_;
  const size_t L0  = E0 + 2 * BNK;                // logprobs offset
  for (int rr = 0; rr < 2; ++rr) {
    const int rx = wave * 2 + rr;
    float cv[8]; int ci[8];
    #pragma unroll
    for (int s = 0; s < 8; ++s) { cv[s] = mv[rx][lane * 8 + s]; ci[s] = mi[rx][lane * 8 + s]; }
    for (int k = 0; k < K_; ++k) {
      float bv = cv[0]; int bi = ci[0];
      #pragma unroll
      for (int s = 1; s < 8; ++s)
        if (cv[s] > bv || (cv[s] == bv && ci[s] < bi)) { bv = cv[s]; bi = ci[s]; }
      #pragma unroll
      for (int off = 16; off > 0; off >>= 1) {
        const float ov = __shfl_xor(bv, off, 32);
        const int   oi = __shfl_xor(bi, off, 32);
        if (ov > bv || (ov == bv && oi < bi)) { bv = ov; bi = oi; }
      }
      #pragma unroll
      for (int s = 0; s < 8; ++s) if (ci[s] == bi) cv[s] = -3.4e38f;  // retire winner
      if (lane == 0) {
        const size_t e = (size_t)b * N_ * K_ + (size_t)(i0 + rx) * K_ + k;
        out[L0 + e]       = bv;                          // logprobs
        out[E0 + e]       = (float)(bi + b * N_);        // edges[0] = col idx + b*N
        out[E0 + BNK + e] = (float)(i0 + rx + b * N_);   // edges[1] = row idx + b*N
      }
    }
  }
}

extern "C" void kernel_launch(void* const* d_in, const int* in_sizes, int n_in,
                              void* d_out, int out_size, void* d_ws, size_t ws_size,
                              hipStream_t stream) {
  (void)in_sizes; (void)n_in; (void)out_size; (void)ws_size;
  const float* x    = (const float*)d_in[0];
  // d_in[1] = A (unused, identity embed)
  const float* temp = (const float*)d_in[2];
  const float* q    = (const float*)d_in[3];
  float* out = (float*)d_out;

  float*    wsf  = (float*)d_ws;
  float*    mean = wsf;                       // B*D        = 512 floats
  float*    sq   = wsf + B_ * D_;             // B*N        = 16384 floats
  _Float16* xch  = (_Float16*)(wsf + B_ * D_ + B_ * N_);  // B*N*D f16 = 4 MB

  // xe output = x (identity embed)
  hipMemcpyAsync(d_out, d_in[0], sizeof(float) * (size_t)B_ * N_ * D_,
                 hipMemcpyDeviceToDevice, stream);

  k_mean  <<<B_,            D_,  0, stream>>>(x, mean);
  k_center<<<B_ * N_,       D_,  0, stream>>>(x, mean, xch, sq);
  k_main  <<<B_ * (N_ / TILE), WAVES * 32, 0, stream>>>(xch, sq, q, temp, out);
}
